// GRUCellModel_82995948028058
// MI455X (gfx1250) — matrix operand order repaired and checked by reading, not compiled, not run on hardware
//
#include <hip/hip_runtime.h>

// ---------------------------------------------------------------------------
// GRU first-step (h0 == 0) fused GEMM + activation for MI455X (gfx1250).
//   out[b,t,h] = (1 - z) * n, with
//   r = sigmoid(gi_r + b_hh_r), z = sigmoid(gi_z + b_hh_z),
//   n = tanh(gi_n + r * b_hh_n),  gi = X @ W_ih^T + b_ih.
// W_hh is mathematically unused (h0 == 0  =>  gh == b_hh).
// bf16 WMMA (v_wmma_f32_16x16x32_bf16), fp32 accumulation, double-buffered
// LDS pipeline, async global->LDS staging of preconverted bf16 weights,
// v_cvt_pk_bf16_f32 packing, non-temporal result stores.
// ---------------------------------------------------------------------------

typedef __attribute__((ext_vector_type(16))) __bf16        v16bf;
typedef __attribute__((ext_vector_type(8)))  float         v8f;
typedef __attribute__((ext_vector_type(4)))  unsigned int  u32x4;
typedef __attribute__((ext_vector_type(2)))  unsigned int  u32x2;
typedef __attribute__((ext_vector_type(4)))  float         f32x4;

#define T_DIM 128
#define B_DIM 256
#define I_DIM 512
#define H_DIM 1024
#define M_DIM (T_DIM * B_DIM)   // 32768 rows of X

#define BM 128     // rows per block
#define BN 64      // h-columns per block (per gate; block covers 3 gates)
#define BK 32      // K step (bf16 WMMA K)
#define NK (I_DIM / BK)
#define LDA 40     // padded LDS row stride in ushorts (80B, conflict-free b128)
#define LDB 40

union Frag {
  u32x4 q[2];   // 32 bytes
  v16bf v;      // 16 bf16
};

__device__ __forceinline__ unsigned bf16pk(float a, float b) {
#if defined(__gfx1250__)
  // dst.lo = bf16(a), dst.hi = bf16(b), round-to-nearest-even
  unsigned r;
  asm("v_cvt_pk_bf16_f32 %0, %1, %2" : "=v"(r) : "v"(a), "v"(b));
  return r;
#else
  // host pass / other arch: manual RNE pack (a low 16, b high 16)
  unsigned ua = __float_as_uint(a);
  unsigned ub = __float_as_uint(b);
  ua += 0x7FFFu + ((ua >> 16) & 1u);
  ub += 0x7FFFu + ((ub >> 16) & 1u);
  return (ua >> 16) | (ub & 0xFFFF0000u);
#endif
}

__device__ __forceinline__ float sigmoid_f(float x) {
  return 1.0f / (1.0f + __expf(-x));
}
__device__ __forceinline__ float tanh_f(float x) {
  float e = __expf(-2.0f * x);
  return (1.0f - e) / (1.0f + e);
}

// LDS byte offset of a generic shared-memory pointer (low 32 bits of the
// generic address are the wave-relative LDS offset on CDNA5).
__device__ __forceinline__ unsigned lds_off(const void* p) {
  return (unsigned)(size_t)p;
}

// W_ih fp32 -> bf16 one-time converter (runs each launch; ~9 MB of traffic).
__global__ void __launch_bounds__(256)
wih_to_bf16(const float* __restrict__ w, unsigned short* __restrict__ o) {
  const int i = blockIdx.x * 256 + threadIdx.x;   // one float4 per thread
  const f32x4 f = reinterpret_cast<const f32x4*>(w)[i];
  u32x2 p;
  p.x = bf16pk(f.x, f.y);
  p.y = bf16pk(f.z, f.w);
  reinterpret_cast<u32x2*>(o)[i] = p;
}

template <bool PRE_BF16>
__global__ void __launch_bounds__(256)
gru_step_fused(const float* __restrict__ x,               // (T*B, I) row-major
               const float* __restrict__ w_ih,            // (3H, I) fp32
               const unsigned short* __restrict__ wbf,    // (3H, I) bf16 (ws)
               const float* __restrict__ b_ih,            // (3H)
               const float* __restrict__ b_hh,            // (3H)
               float* __restrict__ out)                   // (B, T, H)
{
  __shared__ __align__(16) unsigned short Als[2][BM * LDA];       // 2x10 KB
  __shared__ __align__(16) unsigned short Bls[2][3 * BN * LDB];   // 2x15 KB

  const int tid  = threadIdx.x;
  const int lane = tid & 31;
  const int wave = tid >> 5;      // 0..7
  const int wm   = wave & 3;      // 4 row groups of 32
  const int wn   = wave >> 2;     // 2 col groups of 32
  const int half = lane >> 4;     // 0/1 (ISA lane-half for fragments)
  const int lidx = lane & 15;

  const int n0 = blockIdx.x * BN;   // h-column tile base (per gate)
  const int m0 = blockIdx.y * BM;   // row tile base (m = t*B + b)

  v8f acc[3][2][2] = {};            // [gate][mi][ni], fp32 16x16 accumulators

  const float* aBase = x + (size_t)m0 * I_DIM;

  // --- B tile staging: bf16 copy via async global->LDS, or fp32+convert ---
  auto stageB_async = [&](int k, int buf) {
#pragma unroll
    for (int j = 0; j < 3; ++j) {
      const int idx  = tid + 256 * j;   // 0..767 (16B transfers)
      const int bRow = idx >> 2;        // 0..191
      const int seg  = idx & 3;
      const int gate = bRow >> 6;
      const int nl   = bRow & 63;
      const unsigned short* src =
          wbf + (size_t)(gate * H_DIM + n0 + nl) * I_DIM + k + seg * 8;
      const unsigned dst = lds_off(&Bls[buf][bRow * LDB + seg * 8]);
      asm volatile("global_load_async_to_lds_b128 %0, %1, off"
                   :: "v"(dst), "v"(src) : "memory");
    }
  };
  auto asyncB_wait = [&]() {
    asm volatile("s_wait_asynccnt 0x0" ::: "memory");
  };

  // ---- prologue: stage k-tile 0 into buffer 0 ----
  {
#pragma unroll
    for (int j = 0; j < 4; ++j) {
      const int idx = tid + 256 * j;   // 0..1023
      const int row = idx >> 3;        // 0..127
      const int seg = idx & 7;
      const f32x4 f = *reinterpret_cast<const f32x4*>(
          aBase + (size_t)row * I_DIM + seg * 4);
      u32x2 p;
      p.x = bf16pk(f.x, f.y);
      p.y = bf16pk(f.z, f.w);
      *reinterpret_cast<u32x2*>(&Als[0][row * LDA + seg * 4]) = p;
    }
    if (PRE_BF16) {
      stageB_async(0, 0);
      asyncB_wait();
    } else {
#pragma unroll
      for (int j = 0; j < 6; ++j) {
        const int idx  = tid + 256 * j;  // 0..1535
        const int bRow = idx >> 3;
        const int seg  = idx & 7;
        const int gate = bRow >> 6;
        const int nl   = bRow & 63;
        const f32x4 f = *reinterpret_cast<const f32x4*>(
            w_ih + (size_t)(gate * H_DIM + n0 + nl) * I_DIM + seg * 4);
        u32x2 p;
        p.x = bf16pk(f.x, f.y);
        p.y = bf16pk(f.z, f.w);
        *reinterpret_cast<u32x2*>(&Bls[0][bRow * LDB + seg * 4]) = p;
      }
    }
    __syncthreads();
  }

  // ---- main loop: compute buffer `cur` while staging `nxt` ----
  for (int kt = 0; kt < NK; ++kt) {
    const int cur = kt & 1;
    const int nxt = cur ^ 1;
    const bool haveNext = (kt + 1 < NK);
    const int k1 = (kt + 1) * BK;

    // (1) issue next-tile global loads first so latency hides under WMMA
    f32x4 aReg[4];
    f32x4 bReg[6];
    if (haveNext) {
#pragma unroll
      for (int j = 0; j < 4; ++j) {
        const int idx = tid + 256 * j;
        const int row = idx >> 3;
        const int seg = idx & 7;
        aReg[j] = *reinterpret_cast<const f32x4*>(
            aBase + (size_t)row * I_DIM + k1 + seg * 4);
      }
      if (PRE_BF16) {
        stageB_async(k1, nxt);       // async: goes straight to LDS
      } else {
#pragma unroll
        for (int j = 0; j < 6; ++j) {
          const int idx  = tid + 256 * j;
          const int bRow = idx >> 3;
          const int seg  = idx & 7;
          const int gate = bRow >> 6;
          const int nl   = bRow & 63;
          bReg[j] = *reinterpret_cast<const f32x4*>(
              w_ih + (size_t)(gate * H_DIM + n0 + nl) * I_DIM + k1 + seg * 4);
        }
      }
      if (kt + 2 < NK) {   // touch k+2 A slice into cache
        __builtin_prefetch(aBase + (size_t)(tid >> 1) * I_DIM + k1 + BK, 0, 1);
      }
    }

    // (2) compute from `cur`
    // A 16x32: lanes 0-15 M=0..15; VGPR0-3: K=8h..8h+7, VGPR4-7: K=16+8h..
    Frag af[2];
#pragma unroll
    for (int mi = 0; mi < 2; ++mi) {
      const int rowm = wm * 32 + mi * 16 + lidx;
      const unsigned short* p = &Als[cur][rowm * LDA + 8 * half];
      af[mi].q[0] = *reinterpret_cast<const u32x4*>(p);
      af[mi].q[1] = *reinterpret_cast<const u32x4*>(p + 16);
    }
    // B 32x16: lanes hold N=0..15; VGPR0-3: K=8h..8h+7, VGPR4-7: K=16+8h..
    Frag bfr[3][2];
#pragma unroll
    for (int g = 0; g < 3; ++g) {
#pragma unroll
      for (int ni = 0; ni < 2; ++ni) {
        const int nl = wn * 32 + ni * 16 + lidx;
        const unsigned short* p = &Bls[cur][(g * BN + nl) * LDB + 8 * half];
        bfr[g][ni].q[0] = *reinterpret_cast<const u32x4*>(p);
        bfr[g][ni].q[1] = *reinterpret_cast<const u32x4*>(p + 16);
      }
    }
#pragma unroll
    for (int g = 0; g < 3; ++g) {
#pragma unroll
      for (int mi = 0; mi < 2; ++mi) {
#pragma unroll
        for (int ni = 0; ni < 2; ++ni) {
          acc[g][mi][ni] = __builtin_amdgcn_wmma_f32_16x16x32_bf16(
              false, af[mi].v, false, bfr[g][ni].v,
              (short)0, acc[g][mi][ni], false, false);
        }
      }
    }

    // (3) convert + store next tile into `nxt`, fence async, barrier
    if (haveNext) {
#pragma unroll
      for (int j = 0; j < 4; ++j) {
        const int idx = tid + 256 * j;
        const int row = idx >> 3;
        const int seg = idx & 7;
        u32x2 p;
        p.x = bf16pk(aReg[j].x, aReg[j].y);
        p.y = bf16pk(aReg[j].z, aReg[j].w);
        *reinterpret_cast<u32x2*>(&Als[nxt][row * LDA + seg * 4]) = p;
      }
      if (!PRE_BF16) {
#pragma unroll
        for (int j = 0; j < 6; ++j) {
          const int idx  = tid + 256 * j;
          const int bRow = idx >> 3;
          const int seg  = idx & 7;
          u32x2 p;
          p.x = bf16pk(bReg[j].x, bReg[j].y);
          p.y = bf16pk(bReg[j].z, bReg[j].w);
          *reinterpret_cast<u32x2*>(&Bls[nxt][bRow * LDB + seg * 4]) = p;
        }
      } else {
        asyncB_wait();
      }
    }
    __syncthreads();
  }

  // ---- fused GRU epilogue (h_prev == 0  =>  gh == b_hh) ----
  // C/D layout: lane (half,lidx), VGPR r: M = 8*half + r, N = lidx.
  // Output is write-once/streamed: non-temporal stores keep L2 for A reuse.
#pragma unroll
  for (int ni = 0; ni < 2; ++ni) {
    const int n  = n0 + wn * 32 + ni * 16 + lidx;
    const float br = b_ih[n]             + b_hh[n];
    const float bz = b_ih[H_DIM + n]     + b_hh[H_DIM + n];
    const float bn = b_ih[2 * H_DIM + n];
    const float hn = b_hh[2 * H_DIM + n];
#pragma unroll
    for (int mi = 0; mi < 2; ++mi) {
#pragma unroll
      for (int r = 0; r < 8; ++r) {
        const int m = m0 + wm * 32 + mi * 16 + half * 8 + r;  // m = t*B + b
        const int t = m >> 8;      // / B_DIM
        const int b = m & 255;     // % B_DIM
        const float rg = sigmoid_f(acc[0][mi][ni][r] + br);
        const float zg = sigmoid_f(acc[1][mi][ni][r] + bz);
        const float ng = tanh_f(acc[2][mi][ni][r] + bn + rg * hn);
        __builtin_nontemporal_store(
            (1.0f - zg) * ng,
            out + (size_t)b * (T_DIM * H_DIM) + (size_t)t * H_DIM + n);
      }
    }
  }
}

extern "C" void kernel_launch(void* const* d_in, const int* in_sizes, int n_in,
                              void* d_out, int out_size, void* d_ws, size_t ws_size,
                              hipStream_t stream) {
  (void)in_sizes; (void)n_in; (void)out_size;
  const float* x    = (const float*)d_in[0];  // (T,B,I)
  const float* w_ih = (const float*)d_in[1];  // (3H,I)
  // d_in[2] = W_hh: unused (h0 == 0 makes gh == b_hh)
  const float* b_ih = (const float*)d_in[3];
  const float* b_hh = (const float*)d_in[4];
  float* out = (float*)d_out;                 // (B,T,H)

  dim3 grid(H_DIM / BN, M_DIM / BM);          // (16, 256): n-fastest => A reuse in L2

  const size_t wbytes = (size_t)3 * H_DIM * I_DIM * sizeof(unsigned short); // 3 MB
  if (ws_size >= wbytes) {
    unsigned short* wbf = (unsigned short*)d_ws;
    const int n4 = (3 * H_DIM * I_DIM) / 4;   // float4s to convert
    wih_to_bf16<<<n4 / 256, 256, 0, stream>>>(w_ih, wbf);
    gru_step_fused<true><<<grid, 256, 0, stream>>>(x, w_ih, wbf, b_ih, b_hh, out);
  } else {
    gru_step_fused<false><<<grid, 256, 0, stream>>>(x, w_ih, nullptr, b_ih, b_hh, out);
  }
}
